// PhiSoftMax_28750511079450
// MI455X (gfx1250) — matrix operand order repaired and checked by reading, not compiled, not run on hardware
//
#include <hip/hip_runtime.h>
#include <math.h>

// ---- problem constants (from reference) ----
#define B_ 2
#define L_ 2048
#define S_ 2048
#define H_ 8
#define E_ 64
#define WAVES_PER_BLOCK 8
#define NEGBIG (-1.0e30f)

typedef __attribute__((ext_vector_type(16))) __bf16 v16bf;
typedef __attribute__((ext_vector_type(8)))  float  v8f;

static __device__ __forceinline__ __bf16 f2bf(float f) {
  return static_cast<__bf16>(f);   // native v_cvt (RNE) on gfx1250
}

__global__ __launch_bounds__(256) void phisoftmax_attn_kernel(
    const float* __restrict__ q, const float* __restrict__ kmat,
    const float* __restrict__ vmat, const int* __restrict__ pos,
    const int* __restrict__ causal_flag, const int* __restrict__ hard,
    const float* __restrict__ phi, const float* __restrict__ log_tau,
    const float* __restrict__ thr_p, const float* __restrict__ u_,
    float* __restrict__ out)
{
  // per-wave P staging buffer: 16x32 bf16 (1KB per wave)
  __shared__ __align__(16) __bf16 pshm[WAVES_PER_BLOCK][16 * 32];

  const int tid  = threadIdx.x;
  const int wave = tid >> 5;
  const int lane = tid & 31;
  const int lo   = lane & 15;   // N index within 16x16 fragments
  const int hi   = lane >> 4;   // half-wave selector

  const int wid = blockIdx.x * WAVES_PER_BLOCK + wave;   // 0..2047
  const int lt  = wid & 127;            // L/16 = 128 row tiles
  const int h   = (wid >> 7) & (H_ - 1);
  const int b   = wid >> 10;
  const int l0  = lt << 4;

  const float scale   = 0.125f;  // 1/sqrt(64)
  const float tau     = fminf(fmaxf(__expf(log_tau[0]), 0.1f), 5.0f);
  const float inv_tau = 1.0f / tau;
  const float thr     = fminf(fmaxf(thr_p[0], 0.01f), 0.99f);
  const int   causal  = causal_flag[0];

  // ---- Q A-fragments (16x32 bf16 layout), E split into [0,32) and [32,64) ----
  v16bf qa0, qa1;
  {
    const float* qrow = q + ((size_t)((b * L_ + l0 + lo) * H_ + h)) * E_;
    const int eb0 = hi * 16;
    const int eb1 = 32 + hi * 16;
#pragma unroll
    for (int e = 0; e < 16; ++e) {
      qa0[e] = f2bf(qrow[eb0 + e]);
      qa1[e] = f2bf(qrow[eb1 + e]);
    }
  }

  int pq[8];
#pragma unroll
  for (int r = 0; r < 8; ++r) pq[r] = pos[b * L_ + l0 + r + hi * 8];

  float rowmax[8], rowsum[8];
#pragma unroll
  for (int r = 0; r < 8; ++r) { rowmax[r] = NEGBIG; rowsum[r] = 0.0f; }

  v8f o0 = {}, o1 = {}, o2 = {}, o3 = {};

  const int s_end = causal ? (l0 + 16) : S_;   // tiles fully above diagonal skipped
  for (int s0 = 0; s0 < s_end; s0 += 32) {
    // ---------------- scores: two 16x16 tiles via WMMA ----------------
    v8f c0 = {}, c1 = {};
    {
      const float* kr0 = kmat + ((size_t)((b * S_ + s0 + lo) * H_ + h)) * E_ + hi * 16;
      const float* kr1 = kmat + ((size_t)((b * S_ + s0 + 16 + lo) * H_ + h)) * E_ + hi * 16;
      v16bf kb;
#pragma unroll
      for (int e = 0; e < 16; ++e) kb[e] = f2bf(kr0[e]);
      c0 = __builtin_amdgcn_wmma_f32_16x16x32_bf16(false, qa0, false, kb, (short)0, c0, false, false);
#pragma unroll
      for (int e = 0; e < 16; ++e) kb[e] = f2bf(kr0[32 + e]);
      c0 = __builtin_amdgcn_wmma_f32_16x16x32_bf16(false, qa1, false, kb, (short)0, c0, false, false);
#pragma unroll
      for (int e = 0; e < 16; ++e) kb[e] = f2bf(kr1[e]);
      c1 = __builtin_amdgcn_wmma_f32_16x16x32_bf16(false, qa0, false, kb, (short)0, c1, false, false);
#pragma unroll
      for (int e = 0; e < 16; ++e) kb[e] = f2bf(kr1[32 + e]);
      c1 = __builtin_amdgcn_wmma_f32_16x16x32_bf16(false, qa1, false, kb, (short)0, c1, false, false);
    }

    int pk0 = 0, pk1 = 0;
    if (causal) {
      pk0 = pos[b * L_ + s0 + lo];
      pk1 = pos[b * L_ + s0 + 16 + lo];
    }

    // prefetch next tile's mask-tensor lines (global_prefetch_b8)
    if (s0 + 32 < s_end) {
      size_t nb = ((size_t)h * L_ + (size_t)(l0 + hi * 8)) * S_ + (size_t)(s0 + 32 + lo);
      __builtin_prefetch(&phi[nb], 0, 1);
      __builtin_prefetch(&u_[nb], 0, 1);
    }

    // ---------------- masks + Gumbel-sigmoid penalty ----------------
    float lg0[8], lg1[8];
#pragma unroll
    for (int r = 0; r < 8; ++r) {
      const int li = l0 + r + hi * 8;
      const size_t base = ((size_t)h * L_ + (size_t)li) * S_;
      {
        const int si = s0 + lo;
        float sc = c0[r];
        const float ph = phi[base + si];
        const float uu = u_[base + si];
        const float g  = __logf(uu + 1e-8f) - __logf(1.0f - uu + 1e-8f);
        const float mrel = 1.0f / (1.0f + __expf(-(g + ph) * inv_tau));
        sc += 10000.0f * fminf(mrel - thr, 0.0f);
        if (causal && (pk0 > pq[r])) sc = NEGBIG;
        if (hard[(size_t)li * S_ + si] == 0) sc = NEGBIG;
        lg0[r] = scale * sc;
      }
      {
        const int si = s0 + 16 + lo;
        float sc = c1[r];
        const float ph = phi[base + si];
        const float uu = u_[base + si];
        const float g  = __logf(uu + 1e-8f) - __logf(1.0f - uu + 1e-8f);
        const float mrel = 1.0f / (1.0f + __expf(-(g + ph) * inv_tau));
        sc += 10000.0f * fminf(mrel - thr, 0.0f);
        if (causal && (pk1 > pq[r])) sc = NEGBIG;
        if (hard[(size_t)li * S_ + si] == 0) sc = NEGBIG;
        lg1[r] = scale * sc;
      }
    }

    // ---------------- online softmax update ----------------
#pragma unroll
    for (int r = 0; r < 8; ++r) {
      float tm = fmaxf(lg0[r], lg1[r]);
#pragma unroll
      for (int off = 8; off >= 1; off >>= 1) tm = fmaxf(tm, __shfl_xor(tm, off, 32));
      const float nm   = fmaxf(rowmax[r], tm);
      const float corr = __expf(rowmax[r] - nm);
      const float p0   = __expf(lg0[r] - nm);
      const float p1   = __expf(lg1[r] - nm);
      float ts = p0 + p1;
#pragma unroll
      for (int off = 8; off >= 1; off >>= 1) ts += __shfl_xor(ts, off, 32);
      rowsum[r] = rowsum[r] * corr + ts;
      rowmax[r] = nm;
      o0[r] *= corr; o1[r] *= corr; o2[r] *= corr; o3[r] *= corr;
      // stage P (row m = r + hi*8, cols lo and 16+lo) to LDS as bf16
      pshm[wave][(r + hi * 8) * 32 + lo]      = f2bf(p0);
      pshm[wave][(r + hi * 8) * 32 + 16 + lo] = f2bf(p1);
    }

    // within-wave LDS visibility (lockstep wave32; no block barrier needed)
    asm volatile("s_wait_dscnt 0" ::: "memory");

    // ---------------- P (16x32) A-fragment from LDS ----------------
    v16bf pa;
    {
      const __bf16* pw = pshm[wave];
#pragma unroll
      for (int e = 0; e < 16; ++e) {
        const int kk = hi * 16 + e;
        pa[e] = pw[lo * 32 + kk];
      }
    }

    // ---------------- O += P @ V  (four E-column tiles) ----------------
    {
      const float* vbase = vmat + ((size_t)((b * S_ + s0 + hi * 16) * H_ + h)) * E_ + lo;
      v16bf vb;
#pragma unroll
      for (int e = 0; e < 16; ++e) vb[e] = f2bf(vbase[(size_t)e * (H_ * E_)]);
      o0 = __builtin_amdgcn_wmma_f32_16x16x32_bf16(false, pa, false, vb, (short)0, o0, false, false);
#pragma unroll
      for (int e = 0; e < 16; ++e) vb[e] = f2bf(vbase[(size_t)e * (H_ * E_) + 16]);
      o1 = __builtin_amdgcn_wmma_f32_16x16x32_bf16(false, pa, false, vb, (short)0, o1, false, false);
#pragma unroll
      for (int e = 0; e < 16; ++e) vb[e] = f2bf(vbase[(size_t)e * (H_ * E_) + 32]);
      o2 = __builtin_amdgcn_wmma_f32_16x16x32_bf16(false, pa, false, vb, (short)0, o2, false, false);
#pragma unroll
      for (int e = 0; e < 16; ++e) vb[e] = f2bf(vbase[(size_t)e * (H_ * E_) + 48]);
      o3 = __builtin_amdgcn_wmma_f32_16x16x32_bf16(false, pa, false, vb, (short)0, o3, false, false);
    }
  }

  // ---------------- finalize: divide by rowsum, write (B,L,H,E) ----------------
#pragma unroll
  for (int r = 0; r < 8; ++r) {
    const float inv = 1.0f / rowsum[r];
    const int li = l0 + r + hi * 8;
    float* orow = out + ((size_t)((b * L_ + li) * H_ + h)) * E_ + lo;
    orow[0]  = o0[r] * inv;
    orow[16] = o1[r] * inv;
    orow[32] = o2[r] * inv;
    orow[48] = o3[r] * inv;
  }
}

extern "C" void kernel_launch(void* const* d_in, const int* in_sizes, int n_in,
                              void* d_out, int out_size, void* d_ws, size_t ws_size,
                              hipStream_t stream) {
  (void)in_sizes; (void)n_in; (void)out_size; (void)d_ws; (void)ws_size;
  const float* q       = (const float*)d_in[0];
  const float* k       = (const float*)d_in[1];
  const float* v       = (const float*)d_in[2];
  // d_in[3]=mask_miss_k, d_in[4]=mask_miss_q : unused in forward
  const int*   pos     = (const int*)d_in[5];
  const int*   causal  = (const int*)d_in[6];
  const int*   hard    = (const int*)d_in[7];
  const float* phi     = (const float*)d_in[8];
  const float* log_tau = (const float*)d_in[9];
  const float* thr     = (const float*)d_in[10];
  const float* u       = (const float*)d_in[11];
  float* out = (float*)d_out;

  const int total_waves = B_ * H_ * (L_ / 16);            // 2048
  const int blocks = total_waves / WAVES_PER_BLOCK;       // 256
  hipLaunchKernelGGL(phisoftmax_attn_kernel, dim3(blocks), dim3(32 * WAVES_PER_BLOCK),
                     0, stream, q, k, v, pos, causal, hard, phi, log_tau, thr, u, out);
}
